// FullNeighborhoodMultGraphLayer_61418032333444
// MI455X (gfx1250) — matrix operand order repaired and checked by reading, not compiled, not run on hardware
//
#include <hip/hip_runtime.h>
#include <hip/hip_fp16.h>

typedef __attribute__((ext_vector_type(16))) _Float16 v16h;
typedef __attribute__((ext_vector_type(8)))  float    v8f;

#define N_NODES 8192
#define TPB   512              // 16 waves
#define NWAVE (TPB / 32)
#define TILES 2                // 16-col N-tiles per wave (16 waves * 2 = 32 tiles)
#define NPB   4                // nodes per block: each B fragment feeds 4 WMMA chains
#define DD 10
#define F 512
#define LRELU_ALPHA 0.2f

// LDS strides chosen for bank-conflict-free WMMA fragment reads
#define ASTR (F + 8)   // f16 elements: 1040 B/row -> bank step 4/row, 16 rows distinct
#define QSTR (F + 4)   // f32 elements

// ---------------------------------------------------------------------------
// Pack W (512x512 f32) into per-lane WMMA B-fragment order, f16.
// Fragment (tile j, K-step s): lane reads 16 contiguous halves (32 bytes) at
//   byte offset lane*32 + j*16384 + s*1024 from pw.
// B-tile layout: lane = half*16 + n15, element rr -> K = s*32 + 8*half + rr (rr<8), s*32 + 16 + 8*half + (rr-8) (rr>=8).
// ---------------------------------------------------------------------------
__global__ __launch_bounds__(256) void pack_W_kernel(const float* __restrict__ W,
                                                     _Float16* __restrict__ pw) {
  int idx = blockIdx.x * blockDim.x + threadIdx.x;   // 0 .. 512*512-1
  int k = idx >> 9;          // row of W
  int n = idx & 511;         // col of W
  int j  = n >> 4, nn = n & 15;
  int s  = k >> 5, r = k & 31;
  int half = (r >> 3) & 1, rr = (r & 7) | ((r >> 4) << 3);
  int o = (((j * 16 + s) * 32) + half * 16 + nn) * 16 + rr;
  pw[o] = (_Float16)W[idx];
}

__device__ __forceinline__ v16h load_frag16(const _Float16* p) {
  v16h r;
#pragma unroll
  for (int i = 0; i < 16; ++i) r[i] = p[i];
  return r;
}

__device__ __forceinline__ v16h load_afrag(const _Float16* ap) {
  v16h a;
#pragma unroll
  for (int i = 0; i < 8; ++i) a[i]     = ap[i];
#pragma unroll
  for (int i = 0; i < 8; ++i) a[8 + i] = ap[16 + i];
  return a;
}

// ---------------------------------------------------------------------------
// Fully fused layer, NPB=4 nodes per 512-thread block (16 waves).
// ---------------------------------------------------------------------------
__global__ __launch_bounds__(TPB) void fused_node_kernel(
    const float* __restrict__ h,        // [N, F]
    const int*  __restrict__ dst,       // [N*DD]
    const _Float16* __restrict__ pw,    // packed W fragments (512 KB, L2-resident)
    float* __restrict__ out)            // [N, DD, F]
{
  __shared__ _Float16 Ald[NPB][16 * ASTR]; // neighbor features, f16, rows 10..15 = 0
  __shared__ float    qld[16 * QSTR];      // q of one node at a time, f32
  __shared__ float    sc[16 * 16];         // leaky-relu'd scores
  __shared__ float    att[16 * 16];        // softmax over d per column e

  const int n0   = blockIdx.x * NPB;
  const int tid  = threadIdx.x;
  const int lane = tid & 31;
  const int w    = tid >> 5;            // wave id 0..15
  const int half = lane >> 4;
  const int l15  = lane & 15;

  // Per-lane fragment base (halves); all fragment addrs are immediate offsets.
  const _Float16* wb = pw + (size_t)lane * 16 + (size_t)w * (TILES * 8192);

  // Warm the cache for the first K-steps while we stage the gather.
#pragma unroll
  for (int t = 0; t < TILES; ++t) {
    __builtin_prefetch(wb + t * 8192, 0, 3);
    __builtin_prefetch(wb + t * 8192 + 512, 0, 3);
  }

  // ---- Stage A: gather h[dst] rows for all NPB nodes (zero-pad rows 10..15) -
  {
    const int row = tid >> 3;           // 64 groups of 8 threads, one row each
    const int l8  = tid & 7;
    const int v   = row >> 4;           // node within group
    const int r   = row & 15;           // row within node's A
    _Float16* arow = &Ald[v][r * ASTR];
    if (r < DD) {
      const int srcRow = dst[(n0 + v) * DD + r];
      const float4* hv = (const float4*)(h + (size_t)srcRow * F);
#pragma unroll
      for (int i = 0; i < 16; ++i) {
        int c4 = l8 + 8 * i;            // float4 index 0..127
        float4 x = hv[c4];
        int c = c4 * 4;
        arow[c + 0] = (_Float16)x.x; arow[c + 1] = (_Float16)x.y;
        arow[c + 2] = (_Float16)x.z; arow[c + 3] = (_Float16)x.w;
      }
    } else {
#pragma unroll
      for (int i = 0; i < 16; ++i) {
        int c = (l8 + 8 * i) * 4;
        arow[c + 0] = (_Float16)0.f; arow[c + 1] = (_Float16)0.f;
        arow[c + 2] = (_Float16)0.f; arow[c + 3] = (_Float16)0.f;
      }
    }
  }
  __syncthreads();

  // ---- GEMM: q[v] = A[v](16x512) @ W; wave w owns cols [w*32, w*32+32) ------
  const _Float16* arowv[NPB];
#pragma unroll
  for (int v = 0; v < NPB; ++v) arowv[v] = &Ald[v][l15 * ASTR + half * 8];

  v8f acc[NPB][TILES] = {};             // per node, per tile: 16x16 f32

  // Pipelines: B at distance 2 (L2 round-trips), A at distance 1 (LDS), so the
  // WMMA block of step s has no same-step load dependences at all.
  v16h b0[TILES], b1[TILES];
#pragma unroll
  for (int t = 0; t < TILES; ++t) b0[t] = load_frag16(wb + t * 8192);
#pragma unroll
  for (int t = 0; t < TILES; ++t) b1[t] = load_frag16(wb + t * 8192 + 512);

  v16h av[NPB];
#pragma unroll
  for (int v = 0; v < NPB; ++v) av[v] = load_afrag(arowv[v]);

#pragma unroll
  for (int s = 0; s < 16; ++s) {
    // prefetch B for step s+2 (longest latency: issue first)
    v16h bn[TILES];
    if (s < 14) {
#pragma unroll
      for (int t = 0; t < TILES; ++t)
        bn[t] = load_frag16(wb + t * 8192 + (s + 2) * 512);
    } else {
#pragma unroll
      for (int t = 0; t < TILES; ++t) bn[t] = b1[t];
    }

    // prefetch A for step s+1
    v16h avn[NPB];
    if (s < 15) {
#pragma unroll
      for (int v = 0; v < NPB; ++v) avn[v] = load_afrag(arowv[v] + (s + 1) * 32);
    } else {
#pragma unroll
      for (int v = 0; v < NPB; ++v) avn[v] = av[v];
    }

#pragma unroll
    for (int v = 0; v < NPB; ++v)
#pragma unroll
      for (int t = 0; t < TILES; ++t)
        acc[v][t] = __builtin_amdgcn_wmma_f32_16x16x32_f16(
            false, av[v], false, b0[t], (short)0, acc[v][t], false, false);

#pragma unroll
    for (int v = 0; v < NPB; ++v) av[v] = avn[v];
#pragma unroll
    for (int t = 0; t < TILES; ++t) { b0[t] = b1[t]; b1[t] = bn[t]; }
  }

  // ---- Epilogue per node (reuse one qld buffer) -----------------------------
#pragma unroll
  for (int v = 0; v < NPB; ++v) {
    // Spill q tiles to LDS (C layout: lane->n, VGPR r -> m = r + 8*half)
#pragma unroll
    for (int t = 0; t < TILES; ++t) {
      const int col = (w * TILES + t) * 16 + l15;
#pragma unroll
      for (int r = 0; r < 8; ++r) {
        qld[(r + 8 * half) * QSTR + col] = acc[v][t][r];
      }
    }
    __syncthreads();

    // scores[d][e] = dot(q[d], q[e]) in f32, then LeakyReLU
    for (int p = w; p < DD * DD; p += NWAVE) {
      const int d = p / DD, e = p % DD;
      const float* qd = &qld[d * QSTR];
      const float* qe = &qld[e * QSTR];
      float accp = 0.f;
#pragma unroll
      for (int i = 0; i < 16; ++i)
        accp = fmaf(qd[lane * 16 + i], qe[lane * 16 + i], accp);
#pragma unroll
      for (int off = 16; off > 0; off >>= 1)
        accp += __shfl_xor(accp, off, 32);
      if (lane == 0)
        sc[d * 16 + e] = accp > 0.f ? accp : LRELU_ALPHA * accp;
    }
    __syncthreads();

    // softmax over d (axis=1) per column e
    if (tid < DD) {
      const int e = tid;
      float mx = -3.0e38f;
      for (int d = 0; d < DD; ++d) mx = fmaxf(mx, sc[d * 16 + e]);
      float ex[DD];
      float sum = 0.f;
      for (int d = 0; d < DD; ++d) { ex[d] = expf(sc[d * 16 + e] - mx); sum += ex[d]; }
      float inv = 1.f / sum;
      for (int d = 0; d < DD; ++d) att[d * 16 + e] = ex[d] * inv;
    }
    __syncthreads();

    // h'[d][f] = sum_e att[d][e] * q[e][f]; ELU; store (512 threads = 512 cols)
    {
      const int f = tid;
      float qv[DD];
#pragma unroll
      for (int e = 0; e < DD; ++e) qv[e] = qld[e * QSTR + f];
#pragma unroll
      for (int d = 0; d < DD; ++d) {
        float hp = 0.f;
#pragma unroll
        for (int e = 0; e < DD; ++e) hp = fmaf(att[d * 16 + e], qv[e], hp);
        float y = hp > 0.f ? hp : expm1f(hp);
        out[((size_t)(n0 + v) * DD + d) * F + f] = y;
      }
    }
    __syncthreads();   // qld/sc/att reused by next node
  }
}

// ---------------------------------------------------------------------------
extern "C" void kernel_launch(void* const* d_in, const int* in_sizes, int n_in,
                              void* d_out, int out_size, void* d_ws, size_t ws_size,
                              hipStream_t stream) {
  // setup_inputs order: h, edge_feats, adj, src, dst, W, W_edge
  const float* h   = (const float*)d_in[0];
  // d_in[1] edge_feats : unused downstream (We dead)   d_in[2] adj : unused
  // d_in[3] src : implied by layout (src = repeat(arange(N), D))
  const int*   dst = (const int*)d_in[4];
  const float* W   = (const float*)d_in[5];
  // d_in[6] W_edge : unused
  float*    out = (float*)d_out;
  _Float16* pw  = (_Float16*)d_ws;      // 512*512*2 = 512 KB of workspace

  pack_W_kernel<<<(512 * 512) / 256, 256, 0, stream>>>(W, pw);
  fused_node_kernel<<<N_NODES / NPB, TPB, 0, stream>>>(h, dst, pw, out);

  (void)in_sizes; (void)n_in; (void)out_size; (void)ws_size;
}
